// Attention_72464688218562
// MI455X (gfx1250) — 16-bit and 8-bit matrix operand order repaired and checked by reading, compiled, not run on hardware
//
#include <hip/hip_runtime.h>
#include <hip/hip_bf16.h>
#include <cstdint>

typedef __attribute__((ext_vector_type(16))) _Float16 v16h;
typedef __attribute__((ext_vector_type(8)))  float    v8f;
typedef __attribute__((ext_vector_type(8)))  int      v8i;
typedef __attribute__((ext_vector_type(4)))  int      v4i;

#define S_TOK    1024
#define DIM      2560
#define HEAD_DIM 128
#define N_HEADS  20
#define N_KV     5
#define QKV_OUT  3840
#define GS       128
#define NGROUPS  20

__global__ __launch_bounds__(256)
void pack_w_kernel(const float* __restrict__ w, signed char* __restrict__ wi, size_t n) {
    size_t i = (size_t)blockIdx.x * 256u + threadIdx.x;
    if (i < n) wi[i] = (signed char)__float2int_rn(w[i]);
}

__global__ __launch_bounds__(256)
void quant_rows_kernel(const float* __restrict__ x, signed char* __restrict__ q,
                       float* __restrict__ inv_s, int K) {
    const int t = blockIdx.x;
    const float* row = x + (size_t)t * K;
    float m = 0.f;
    for (int i = threadIdx.x; i < K; i += 256) m = fmaxf(m, fabsf(row[i]));
    __shared__ float sh[256];
    sh[threadIdx.x] = m;
    __syncthreads();
    for (int off = 128; off; off >>= 1) {
        if (threadIdx.x < off) sh[threadIdx.x] = fmaxf(sh[threadIdx.x], sh[threadIdx.x + off]);
        __syncthreads();
    }
    const float mx = fmaxf(sh[0], 1e-5f);
    const float s  = 127.f / mx;
    if (threadIdx.x == 0) inv_s[t] = mx * (1.f / 127.f);
    for (int i = threadIdx.x; i < K; i += 256) {
        float v = rintf(row[i] * s);
        v = fminf(fmaxf(v, -128.f), 127.f);
        q[(size_t)t * K + i] = (signed char)v;
    }
}

__global__ __launch_bounds__(256)
void gemm_iu8_kernel(const signed char* __restrict__ qx, const float* __restrict__ inv_s,
                     const signed char* __restrict__ w,  const float* __restrict__ ws,
                     float* __restrict__ out, int N, int K, int NG) {
    const int m0   = blockIdx.y * 16;
    const int wave = threadIdx.x >> 5;
    const int lane = threadIdx.x & 31;
    const int col  = lane & 15;
    const int hi   = lane >> 4;
    const int n0   = blockIdx.x * 256 + wave * 32;

    __shared__ __align__(16) signed char atile[2][16][128];

    float facc0[8] = {0.f,0.f,0.f,0.f,0.f,0.f,0.f,0.f};
    float facc1[8] = {0.f,0.f,0.f,0.f,0.f,0.f,0.f,0.f};
    const signed char* wrow0 = w + (size_t)(n0 + col) * K;
    const signed char* wrow1 = w + (size_t)(n0 + 16 + col) * K;
    const int nKG = K / 128;

    const int sr  = threadIdx.x >> 4;
    const int scb = (threadIdx.x & 15) * 8;
    const signed char* gsrc0 = qx + (size_t)(m0 + sr) * K + scb;

    {
        unsigned lds = (unsigned)(uintptr_t)&atile[0][sr][scb];
        asm volatile("global_load_async_to_lds_b64 %0, %1, off"
                     :: "v"(lds), "v"(gsrc0) : "memory");
    }

    for (int kg = 0; kg < nKG; ++kg) {
        const int buf = kg & 1;
        if (kg + 1 < nKG) {
            unsigned lds = (unsigned)(uintptr_t)&atile[buf ^ 1][sr][scb];
            const signed char* gs = gsrc0 + (size_t)(kg + 1) * 128;
            asm volatile("global_load_async_to_lds_b64 %0, %1, off"
                         :: "v"(lds), "v"(gs) : "memory");
            asm volatile("s_wait_asynccnt 0x1" ::: "memory");
            __builtin_prefetch(wrow0 + (size_t)(kg + 1) * 128, 0, 3);
            __builtin_prefetch(wrow1 + (size_t)(kg + 1) * 128, 0, 3);
        } else {
            asm volatile("s_wait_asynccnt 0x0" ::: "memory");
        }
        __syncthreads();

        v8i iacc0 = {0,0,0,0,0,0,0,0};
        v8i iacc1 = {0,0,0,0,0,0,0,0};
#pragma unroll
        for (int ks = 0; ks < 2; ++ks) {
            const int kb = ks * 64;
            v8i a;
            const int ab = kb + hi * 16;
            a[0] = *(const int*)&atile[buf][col][ab +  0];
            a[1] = *(const int*)&atile[buf][col][ab +  4];
            a[2] = *(const int*)&atile[buf][col][ab +  8];
            a[3] = *(const int*)&atile[buf][col][ab + 12];
            a[4] = *(const int*)&atile[buf][col][ab + 32];
            a[5] = *(const int*)&atile[buf][col][ab + 36];
            a[6] = *(const int*)&atile[buf][col][ab + 40];
            a[7] = *(const int*)&atile[buf][col][ab + 44];
            v8i b0, b1;
            const signed char* wp0 = wrow0 + (size_t)kg * 128 + kb + hi * 16;
            const signed char* wp1 = wrow1 + (size_t)kg * 128 + kb + hi * 16;
            ((v4i*)&b0)[0] = *(const v4i*)(const void*)(wp0);
            ((v4i*)&b0)[1] = *(const v4i*)(const void*)(wp0 + 32);
            ((v4i*)&b1)[0] = *(const v4i*)(const void*)(wp1);
            ((v4i*)&b1)[1] = *(const v4i*)(const void*)(wp1 + 32);
            iacc0 = __builtin_amdgcn_wmma_i32_16x16x64_iu8(true, a, true, b0, iacc0,
                                                           false, false);
            iacc1 = __builtin_amdgcn_wmma_i32_16x16x64_iu8(true, a, true, b1, iacc1,
                                                           false, false);
        }
        const float sc0 = ws[(size_t)(n0 + col) * NG + kg];
        const float sc1 = ws[(size_t)(n0 + 16 + col) * NG + kg];
#pragma unroll
        for (int i = 0; i < 8; ++i) {
            facc0[i] += (float)iacc0[i] * sc0;
            facc1[i] += (float)iacc1[i] * sc1;
        }
        __syncthreads();
    }

#pragma unroll
    for (int i = 0; i < 8; ++i) {
        const int m = m0 + i + hi * 8;
        const float is = inv_s[m];
        out[(size_t)m * N + n0 + col]      = facc0[i] * is;
        out[(size_t)m * N + n0 + 16 + col] = facc1[i] * is;
    }
}

__global__ __launch_bounds__(128)
void norm_rope_kernel(const float* __restrict__ qkv,
                      const float* __restrict__ qnw, const float* __restrict__ knw,
                      _Float16* __restrict__ qf, _Float16* __restrict__ kf,
                      _Float16* __restrict__ vt) {
    const int s = blockIdx.x, slot = blockIdx.y, d = threadIdx.x;
    __shared__ float sh[128];

    if (slot >= 25) {
        const int g = slot - 25;
        const float v = qkv[(size_t)s * QKV_OUT + 3200 + g * HEAD_DIM + d];
        vt[((size_t)g * HEAD_DIM + d) * S_TOK + s] = (_Float16)v;
        return;
    }
    const bool is_q = (slot < 20);
    const int base = is_q ? slot * HEAD_DIM : 2560 + (slot - 20) * HEAD_DIM;
    const float* nw = is_q ? qnw : knw;

    const float v = qkv[(size_t)s * QKV_OUT + base + d];
    sh[d] = v * v;
    __syncthreads();
    for (int off = 64; off; off >>= 1) {
        if (d < off) sh[d] += sh[d + off];
        __syncthreads();
    }
    const float r  = rsqrtf(sh[0] * (1.f / 128.f) + 1e-5f);
    const float nv = v * r * nw[d];
    __syncthreads();
    sh[d] = nv;
    __syncthreads();

    const int   p = d >> 1;
    const float inv_freq = __expf(-((float)(2 * p) * (1.f / 128.f)) * __logf(500000.f));
    float sn, cs;
    __sincosf((float)s * inv_freq, &sn, &cs);
    const float x0 = sh[2 * p], x1 = sh[2 * p + 1];
    const float o = (d & 1) ? (x0 * sn + x1 * cs) : (x0 * cs - x1 * sn);

    if (is_q) qf[((size_t)slot * S_TOK + s) * HEAD_DIM + d] = (_Float16)o;
    else      kf[((size_t)(slot - 20) * S_TOK + s) * HEAD_DIM + d] = (_Float16)o;
}

__global__ __launch_bounds__(256)
void attn_kernel(const _Float16* __restrict__ qf, const _Float16* __restrict__ kf,
                 const _Float16* __restrict__ vt, float* __restrict__ attn_out) {
    const int qb = blockIdx.x, h = blockIdx.y, g = h >> 2;
    const int wave = threadIdx.x >> 5, lane = threadIdx.x & 31;
    const int c = lane & 15, hi = lane >> 4;

    __shared__ __align__(16) _Float16 qt[16][HEAD_DIM];
    __shared__ __align__(16) _Float16 sc[16][S_TOK];
    __shared__ float red[16][16];
    __shared__ float rowmax[16], rowinv[16];

    {
        const float scale = 0.08838834764831845f;
        for (int i = threadIdx.x; i < 16 * HEAD_DIM; i += 256) {
            const int r = i >> 7, d = i & 127;
            qt[r][d] = (_Float16)((float)qf[((size_t)h * S_TOK + qb * 16 + r) * HEAD_DIM + d]
                                  * scale);
        }
    }
    __syncthreads();

    for (int kb = wave; kb <= qb; kb += 8) {
        v8f acc = {};
#pragma unroll
        for (int dc = 0; dc < 4; ++dc) {
            const int d0 = dc * 32;
            v16h a, b;
            ((v4i*)&a)[0] = *(const v4i*)&qt[c][d0 + hi * 8];
            ((v4i*)&a)[1] = *(const v4i*)&qt[c][d0 + 16 + hi * 8];
            const _Float16* kp =
                kf + ((size_t)g * S_TOK + kb * 16 + c) * HEAD_DIM + d0 + hi * 8;
            ((v4i*)&b)[0] = *(const v4i*)(const void*)kp;
            ((v4i*)&b)[1] = *(const v4i*)(const void*)(kp + 16);
            acc = __builtin_amdgcn_wmma_f32_16x16x32_f16(false, a, false, b,
                                                         (short)0, acc, false, false);
        }
#pragma unroll
        for (int i = 0; i < 8; ++i) {
            const int r = i + hi * 8;
            float v = acc[i];
            if (kb == qb && c > r) v = -3.0e4f;
            sc[r][kb * 16 + c] = (_Float16)v;
        }
    }
    __syncthreads();

    const int L  = (qb + 1) * 16;
    const int Lp = (L + 31) & ~31;
    {
        const int r = threadIdx.x >> 4, t = threadIdx.x & 15;
        float m = -3e38f;
        for (int colx = t; colx < L; colx += 16) m = fmaxf(m, (float)sc[r][colx]);
        red[r][t] = m;
        __syncthreads();
        if (t == 0) {
            float mm = red[r][0];
            for (int i = 1; i < 16; ++i) mm = fmaxf(mm, red[r][i]);
            rowmax[r] = mm;
        }
        __syncthreads();
        const float M = rowmax[r];
        float sum = 0.f;
        for (int colx = t; colx < L; colx += 16) {
            const float p = __expf((float)sc[r][colx] - M);
            sc[r][colx] = (_Float16)p;
            sum += p;
        }
        for (int colx = L + t; colx < Lp; colx += 16) sc[r][colx] = (_Float16)0.f;
        red[r][t] = sum;
        __syncthreads();
        if (t == 0) {
            float s2 = 0.f;
            for (int i = 0; i < 16; ++i) s2 += red[r][i];
            rowinv[r] = 1.f / s2;
        }
    }
    __syncthreads();

    {
        const int dt = wave * 16;
        v8f acc = {};
        for (int t0 = 0; t0 < Lp; t0 += 32) {
            v16h a, b;
            ((v4i*)&a)[0] = *(const v4i*)&sc[c][t0 + hi * 8];
            ((v4i*)&a)[1] = *(const v4i*)&sc[c][t0 + 16 + hi * 8];
            const _Float16* vp =
                vt + ((size_t)g * HEAD_DIM + dt + c) * S_TOK + t0 + hi * 8;
            ((v4i*)&b)[0] = *(const v4i*)(const void*)vp;
            ((v4i*)&b)[1] = *(const v4i*)(const void*)(vp + 16);
            acc = __builtin_amdgcn_wmma_f32_16x16x32_f16(false, a, false, b,
                                                         (short)0, acc, false, false);
        }
#pragma unroll
        for (int i = 0; i < 8; ++i) {
            const int r = i + hi * 8;
            attn_out[(size_t)(qb * 16 + r) * DIM + h * HEAD_DIM + dt + c] =
                acc[i] * rowinv[r];
        }
    }
}

extern "C" void kernel_launch(void* const* d_in, const int* in_sizes, int n_in,
                              void* d_out, int out_size, void* d_ws, size_t ws_size,
                              hipStream_t stream) {
    (void)in_sizes; (void)n_in; (void)out_size; (void)ws_size;
    const float* x      = (const float*)d_in[0];
    const float* w_qkv  = (const float*)d_in[1];
    const float* ws_qkv = (const float*)d_in[2];
    const float* w_o    = (const float*)d_in[3];
    const float* ws_o   = (const float*)d_in[4];
    const float* qnw    = (const float*)d_in[5];
    const float* knw    = (const float*)d_in[6];
    float* out = (float*)d_out;

    char* ws = (char*)d_ws;
    signed char* qx1   = (signed char*)(ws + 0);
    float*       is1   = (float*)      (ws + 2621440);
    signed char* wq_i8 = (signed char*)(ws + 2625536);
    signed char* wo_i8 = (signed char*)(ws + 12455936);
    float*       qkv   = (float*)      (ws + 19009536);
    _Float16*    qf    = (_Float16*)   (ws + 34738176);
    _Float16*    kf    = (_Float16*)   (ws + 39981056);
    _Float16*    vt    = (_Float16*)   (ws + 41291776);
    float*       aout  = (float*)      (ws + 42602496);
    signed char* qx2   = (signed char*)(ws + 53088256);
    float*       is2   = (float*)      (ws + 55709696);

    {
        const size_t n1 = (size_t)QKV_OUT * DIM;
        const size_t n2 = (size_t)DIM * DIM;
        pack_w_kernel<<<dim3((unsigned)((n1 + 255) / 256)), 256, 0, stream>>>(w_qkv, wq_i8, n1);
        pack_w_kernel<<<dim3((unsigned)((n2 + 255) / 256)), 256, 0, stream>>>(w_o,  wo_i8, n2);
    }
    quant_rows_kernel<<<dim3(S_TOK), 256, 0, stream>>>(x, qx1, is1, DIM);
    gemm_iu8_kernel<<<dim3(QKV_OUT / 256, S_TOK / 16), 256, 0, stream>>>(
        qx1, is1, wq_i8, ws_qkv, qkv, QKV_OUT, DIM, NGROUPS);
    norm_rope_kernel<<<dim3(S_TOK, 30), 128, 0, stream>>>(qkv, qnw, knw, qf, kf, vt);
    attn_kernel<<<dim3(S_TOK / 16, N_HEADS), 256, 0, stream>>>(qf, kf, vt, aout);
    quant_rows_kernel<<<dim3(S_TOK), 256, 0, stream>>>(aout, qx2, is2, DIM);
    gemm_iu8_kernel<<<dim3(DIM / 256, S_TOK / 16), 256, 0, stream>>>(
        qx2, is2, wo_i8, ws_o, out, DIM, DIM, NGROUPS);
}
